// GraphTCN_11261404250710
// MI455X (gfx1250) — matrix operand order repaired and checked by reading, not compiled, not run on hardware
//
#include <hip/hip_runtime.h>

// GraphTCN forward for gfx1250 (MI455X).
// All MLP matmuls on v_wmma_f32_16x16x32_f16, 16-row tiles per wave, weights
// resident in B-fragments (rows remapped so each lane's A-gather is a
// contiguous 8-float span -> straight-line global_load_b128). Load path is
// fully branchless: indices clamped, unused half-wave lanes load duplicate
// cachelines and zero via selects. Biases preloaded to registers before the
// tile loop. Per-wave 16x48 LDS staging (DS in-order per wave, no barriers).
// segment_sum via native fp32 global atomics (agg tables are L2-resident).

typedef __attribute__((ext_vector_type(16))) _Float16 v16h;
typedef __attribute__((ext_vector_type(8)))  float    v8f;

#define LDS_COLS 48   // hidden 40 padded to 3 N-blocks of 16

__device__ __forceinline__ int lane_id() { return (int)(threadIdx.x & 31); }

__device__ __forceinline__ v8f wmma32(v16h a, v16h b, v8f c) {
  return __builtin_amdgcn_wmma_f32_16x16x32_f16(
      false, a, false, b, (short)0, c, false, false);
}

__device__ __forceinline__ float4 ld4(const float* p) { return *(const float4*)p; }
__device__ __forceinline__ float2 ld2(const float* p) { return *(const float2*)p; }

// Pack a lane's 16 A-slots: l0,l1 = the 8-float "low" run, u0,u1 = "high" run.
__device__ __forceinline__ v16h pack_a(float4 l0, float4 l1, float4 u0, float4 u1) {
  v16h a;
  a[0]=(_Float16)l0.x;  a[1]=(_Float16)l0.y;  a[2]=(_Float16)l0.z;  a[3]=(_Float16)l0.w;
  a[4]=(_Float16)l1.x;  a[5]=(_Float16)l1.y;  a[6]=(_Float16)l1.z;  a[7]=(_Float16)l1.w;
  a[8]=(_Float16)u0.x;  a[9]=(_Float16)u0.y;  a[10]=(_Float16)u0.z; a[11]=(_Float16)u0.w;
  a[12]=(_Float16)u1.x; a[13]=(_Float16)u1.y; a[14]=(_Float16)u1.z; a[15]=(_Float16)u1.w;
  return a;
}

// B fragment with padded-K -> weight-row remap. map(k) returns weight row or -1.
template <typename M>
__device__ __forceinline__ v16h load_b_map(const float* __restrict__ W,
                                           int Ndim, int n0, M map) {
  int lane = lane_id();
  int n = n0 + (lane & 15);
  int kb = (lane >> 4) << 3;
  v16h b;
#pragma unroll
  for (int j = 0; j < 8; ++j) {
#pragma unroll
    for (int p = 0; p < 2; ++p) {
      int r = map(kb + 2 * j + p + (j >= 4 ? 8 : 0));
      float v = (r >= 0 && n < Ndim) ? W[r * Ndim + n] : 0.0f;
      b[2 * j + p] = (_Float16)v;
    }
  }
  return b;
}

__device__ __forceinline__ v16h load_b_hidden(const float* __restrict__ W,
                                              int Ndim, int k0, int n0) {
  return load_b_map(W, Ndim, n0, [=](int k) -> int {
    int kk = k0 + k; return (kk < 40) ? kk : -1;
  });
}

// A fragment from LDS hidden tile, K block 0 (cols 0..31).
__device__ __forceinline__ v16h a_lds_lo(const float* h48) {
  int lane = lane_id(); int m = lane & 15; int hi = lane >> 4;
  const float* row = h48 + m * LDS_COLS;
  int blo = hi * 8, bup = 16 + hi * 8;
  return pack_a(ld4(row + blo), ld4(row + blo + 4),
                ld4(row + bup), ld4(row + bup + 4));
}
// K block 1 (cols 32..47 live; rest zero).
__device__ __forceinline__ v16h a_lds_hi(const float* h48) {
  int lane = lane_id(); int m = lane & 15; int hi = lane >> 4;
  const float* row = h48 + m * LDS_COLS;
  int blo = 32 + hi * 8;
  float4 z = {0.f, 0.f, 0.f, 0.f};
  return pack_a(ld4(row + blo), ld4(row + blo + 4), z, z);
}

// Store 3 C blocks (16x48) to LDS with preloaded per-lane biases + ReLU.
__device__ __forceinline__ void store_hidden40(float* h48, const v8f* c,
                                               const float* bv3) {
  int lane = lane_id();
  int n = lane & 15, hi = lane >> 4;
#pragma unroll
  for (int nb = 0; nb < 3; ++nb) {
    int col = nb * 16 + n;
#pragma unroll
    for (int r = 0; r < 8; ++r) {
      float v = c[nb][r] + bv3[nb];
      h48[(r + 8 * hi) * LDS_COLS + col] = fmaxf(v, 0.0f);
    }
  }
}

__device__ __forceinline__ void layer_first(const v16h* B1, const float* bv3,
                                            float* h48, v16h a) {
  v8f c[3];
#pragma unroll
  for (int nb = 0; nb < 3; ++nb) { v8f z = {}; c[nb] = wmma32(a, B1[nb], z); }
  store_hidden40(h48, c, bv3);
}

__device__ __forceinline__ void layer_mid(const v16h* B, const float* bv3,
                                          float* h48) {
  v16h a0 = a_lds_lo(h48), a1 = a_lds_hi(h48);
  v8f c[3];
#pragma unroll
  for (int nb = 0; nb < 3; ++nb) {
    v8f z = {};
    c[nb] = wmma32(a0, B[nb], z);
    c[nb] = wmma32(a1, B[3 + nb], c[nb]);
  }
  store_hidden40(h48, c, bv3);
}

__device__ __forceinline__ v8f layer_last(const v16h* B2, const float* h48) {
  v16h a0 = a_lds_lo(h48), a1 = a_lds_hi(h48);
  v8f z = {};
  v8f c = wmma32(a0, B2[0], z);
  return wmma32(a1, B2[1], c);
}

// Stage final values (preloaded per-lane bias, optional ReLU) to LDS cols [0,NPAD).
__device__ __forceinline__ void stage_out(float* h48, v8f c, float ob,
                                          int NPAD, int dorelu) {
  int lane = lane_id();
  int n = lane & 15, hi = lane >> 4;
  if (n < NPAD) {
#pragma unroll
    for (int r = 0; r < 8; ++r) {
      float v = c[r] + ob;
      if (dorelu) v = fmaxf(v, 0.0f);
      h48[(r + 8 * hi) * LDS_COLS + n] = v;
    }
  }
}

// Preload per-lane hidden-bias registers (3 cols per lane).
__device__ __forceinline__ void bias3(const float* __restrict__ b, float* bv3) {
  int n = lane_id() & 15;
#pragma unroll
  for (int nb = 0; nb < 3; ++nb) {
    int col = nb * 16 + n;
    bv3[nb] = (col < 40) ? b[col] : 0.0f;
  }
}
__device__ __forceinline__ float bias1(const float* __restrict__ b, int NOUT) {
  int n = lane_id() & 15;
  return (n < NOUT) ? b[n] : 0.0f;
}

__device__ __forceinline__ float sigmoidf(float x) {
  return 1.0f / (1.0f + expf(-x));
}

// ---------------------------------------------------------------- kernels

__global__ void k_zero(float* __restrict__ p, int n) {
  int i = blockIdx.x * blockDim.x + threadIdx.x;
  if (i < n) p[i] = 0.0f;
}

// Node encoder (outer ReLU): x[N,14] -> 40 -> 5; h stride 8, pad cols zeroed.
__global__ void k_node_enc(const float* __restrict__ x,
                           const float* __restrict__ W1, const float* __restrict__ b1,
                           const float* __restrict__ W2, const float* __restrict__ b2,
                           float* __restrict__ h, int nN) {
  __shared__ __align__(16) float lds[8][16 * LDS_COLS];
  int wid = threadIdx.x >> 5, lane = lane_id();
  int m = lane & 15, hi = lane >> 4;
  float sel = hi ? 0.0f : 1.0f;
  float* h48 = lds[wid];
  v16h B1[3], B2[2];
#pragma unroll
  for (int nb = 0; nb < 3; ++nb)
    B1[nb] = load_b_map(W1, 40, nb * 16, [](int k) -> int { return k < 14 ? k : -1; });
  B2[0] = load_b_hidden(W2, 5, 0, 0);
  B2[1] = load_b_hidden(W2, 5, 32, 0);
  float bv1[3]; bias3(b1, bv1);
  float ob = bias1(b2, 5);
  int tiles = (nN + 15) >> 4;
  int gw = blockIdx.x * (blockDim.x >> 5) + wid;
  int nw = gridDim.x * (blockDim.x >> 5);
  float4 z = {0.f, 0.f, 0.f, 0.f};
  for (int t = gw; t < tiles; t += nw) {
    int base = t << 4;
    int node = base + m; if (node >= nN) node = nN - 1;
    const float* xr = x + (size_t)node * 14 + hi * 8;   // hi=0: k0..7, hi=1: k8..13
    float2 p0 = ld2(xr), p1 = ld2(xr + 2), p2 = ld2(xr + 4);
    float2 p3 = ld2(xr + (hi ? 4 : 6));                 // hi lanes: dup load, zeroed
    float4 l0 = {p0.x, p0.y, p1.x, p1.y};
    float4 l1 = {p2.x, p2.y, p3.x * sel, p3.y * sel};
    layer_first(B1, bv1, h48, pack_a(l0, l1, z, z));
    v8f c = layer_last(B2, h48);
    stage_out(h48, c, ob, 8, 1);
    int nd = base + m;
    if (!hi && nd < nN) {
      const float* orow = h48 + m * LDS_COLS;
      *(float4*)&h[(size_t)nd * 8]     = ld4(orow);
      *(float4*)&h[(size_t)nd * 8 + 4] = ld4(orow + 4);
    }
  }
}

// Edge encoder (outer ReLU): padded k0..3 = ea, k4 = w (if withW). -> 40 -> 4.
__global__ void k_edge_enc(const float* __restrict__ ea, const float* __restrict__ wgt,
                           int withW,
                           const float* __restrict__ W1, const float* __restrict__ b1,
                           const float* __restrict__ W2, const float* __restrict__ b2,
                           float* eall, int nE) {
  __shared__ __align__(16) float lds[8][16 * LDS_COLS];
  int wid = threadIdx.x >> 5, lane = lane_id();
  int m = lane & 15, hi = lane >> 4;
  float sel = hi ? 0.0f : 1.0f;
  float* h48 = lds[wid];
  v16h B1[3], B2[2];
#pragma unroll
  for (int nb = 0; nb < 3; ++nb)
    B1[nb] = load_b_map(W1, 40, nb * 16, [withW](int k) -> int {
      if (withW) { if (k < 4) return k + 1; if (k == 4) return 0; return -1; }
      return (k < 4) ? k : -1;
    });
  B2[0] = load_b_hidden(W2, 4, 0, 0);
  B2[1] = load_b_hidden(W2, 4, 32, 0);
  float bv1[3]; bias3(b1, bv1);
  float ob = bias1(b2, 4);
  int tiles = (nE + 15) >> 4;
  int gw = blockIdx.x * (blockDim.x >> 5) + wid;
  int nw = gridDim.x * (blockDim.x >> 5);
  float4 z = {0.f, 0.f, 0.f, 0.f};
  for (int t = gw; t < tiles; t += nw) {
    int base = t << 4;
    int e0 = base + m; bool eok = e0 < nE; if (!eok) e0 = nE - 1;
    float4 ev = ld4(ea + (size_t)e0 * 4);               // hi lanes: dup cacheline
    float wv = withW ? (wgt[e0] * sel) : 0.0f;          // withW is kernel-uniform
    float4 l0 = {ev.x * sel, ev.y * sel, ev.z * sel, ev.w * sel};
    float4 l1 = {wv, 0.f, 0.f, 0.f};
    layer_first(B1, bv1, h48, pack_a(l0, l1, z, z));
    v8f c = layer_last(B2, h48);
    stage_out(h48, c, ob, 4, 1);
    if (!hi && eok)
      *(float4*)&eall[(size_t)e0 * 16] = ld4(h48 + m * LDS_COLS);
  }
}

// Interaction edge model: padded [k0..7]=h[dst], [k8..15]=h[src], [k16..19]=e.
// 14->40->4 (no act); m -> eall colOut, atomic agg[dst].
__global__ void k_edge_layer(const int* __restrict__ ei,
                             const float* __restrict__ h,
                             float* eall, int colIn, int colOut,
                             const float* __restrict__ W1, const float* __restrict__ b1,
                             const float* __restrict__ W2, const float* __restrict__ b2,
                             float* agg, int nE) {
  __shared__ __align__(16) float lds[8][16 * LDS_COLS];
  int wid = threadIdx.x >> 5, lane = lane_id();
  int m = lane & 15, hi = lane >> 4;
  float sel = hi ? 0.0f : 1.0f;
  float* h48 = lds[wid];
  v16h B1[3], B2[2];
#pragma unroll
  for (int nb = 0; nb < 3; ++nb)
    B1[nb] = load_b_map(W1, 40, nb * 16, [](int k) -> int {
      if (k < 5) return k;
      if (k >= 8 && k < 13) return k - 3;
      if (k >= 16 && k < 20) return k - 6;
      return -1;
    });
  B2[0] = load_b_hidden(W2, 4, 0, 0);
  B2[1] = load_b_hidden(W2, 4, 32, 0);
  float bv1[3]; bias3(b1, bv1);
  float ob = bias1(b2, 4);
  int tiles = (nE + 15) >> 4;
  int gw = blockIdx.x * (blockDim.x >> 5) + wid;
  int nw = gridDim.x * (blockDim.x >> 5);
  float4 z = {0.f, 0.f, 0.f, 0.f};
  for (int t = gw; t < tiles; t += nw) {
    int base = t << 4;
    int e0 = base + m; bool eok = e0 < nE; if (!eok) e0 = nE - 1;
    int s = ei[e0];
    int d = ei[(size_t)nE + e0];
    const float* hr = h + (size_t)(hi ? s : d) * 8;
    float4 l0 = ld4(hr), l1 = ld4(hr + 4);
    float4 e4 = ld4(eall + (size_t)e0 * 16 + colIn);    // hi lanes: dup cacheline
    float4 u0 = {e4.x * sel, e4.y * sel, e4.z * sel, e4.w * sel};
    layer_first(B1, bv1, h48, pack_a(l0, l1, u0, z));
    v8f c = layer_last(B2, h48);
    stage_out(h48, c, ob, 4, 0);
    if (!hi && eok) {
      float4 v = ld4(h48 + m * LDS_COLS);
      *(float4*)&eall[(size_t)e0 * 16 + colOut] = v;
      float* ag = agg + (size_t)d * 4;
      unsafeAtomicAdd(ag + 0, v.x);
      unsafeAtomicAdd(ag + 1, v.y);
      unsafeAtomicAdd(ag + 2, v.z);
      unsafeAtomicAdd(ag + 3, v.w);
    }
  }
}

// Interaction node model: padded [k0..7]=h, [k8..11]=agg. 9->40->5.
// h_new = 0.5h + 0.5dh, stored as full 8-float rows (pad stays 0).
__global__ void k_node_layer(const float* __restrict__ hin,
                             const float* __restrict__ agg,
                             const float* __restrict__ W1, const float* __restrict__ b1,
                             const float* __restrict__ W2, const float* __restrict__ b2,
                             float* __restrict__ hout, int nN) {
  __shared__ __align__(16) float lds[8][16 * LDS_COLS];
  int wid = threadIdx.x >> 5, lane = lane_id();
  int m = lane & 15, hi = lane >> 4;
  float sel = hi ? 0.0f : 1.0f;
  float* h48 = lds[wid];
  v16h B1[3], B2[2];
#pragma unroll
  for (int nb = 0; nb < 3; ++nb)
    B1[nb] = load_b_map(W1, 40, nb * 16, [](int k) -> int {
      if (k < 5) return k;
      if (k >= 8 && k < 12) return k - 3;
      return -1;
    });
  B2[0] = load_b_hidden(W2, 5, 0, 0);
  B2[1] = load_b_hidden(W2, 5, 32, 0);
  float bv1[3]; bias3(b1, bv1);
  float ob = bias1(b2, 5);
  int tiles = (nN + 15) >> 4;
  int gw = blockIdx.x * (blockDim.x >> 5) + wid;
  int nw = gridDim.x * (blockDim.x >> 5);
  float4 z = {0.f, 0.f, 0.f, 0.f};
  for (int t = gw; t < tiles; t += nw) {
    int base = t << 4;
    int node = base + m; bool ok = node < nN; if (!ok) node = nN - 1;
    const float* hrow = hin + (size_t)node * 8;
    const float* arow = agg + (size_t)node * 4;
    float4 l0 = ld4(hi ? arow : hrow);
    float4 t1 = ld4(hi ? arow : hrow + 4);              // hi lanes: dup load, zeroed
    float4 l1 = {t1.x * sel, t1.y * sel, t1.z * sel, t1.w * sel};
    layer_first(B1, bv1, h48, pack_a(l0, l1, z, z));
    v8f c = layer_last(B2, h48);
    stage_out(h48, c, ob, 8, 0);
    if (!hi && ok) {
      const float* orow = h48 + m * LDS_COLS;
      float4 o0 = ld4(orow), o1 = ld4(orow + 4);
      float4 r0 = {0.5f * (l0.x + o0.x), 0.5f * (l0.y + o0.y),
                   0.5f * (l0.z + o0.z), 0.5f * (l0.w + o0.w)};
      float4 r1 = {0.5f * (l1.x + o1.x), 0.5f * (l1.y + o1.y),
                   0.5f * (l1.z + o1.z), 0.5f * (l1.w + o1.w)};
      *(float4*)&hout[(size_t)node * 8]     = r0;
      *(float4*)&hout[(size_t)node * 8 + 4] = r1;
    }
  }
}

// P edge model: padded [k0..7]=h[dst], [k8..15]=h[src], [k16..31]=e_hcs(16).
// 26->40->1; atomic into agg1[dst].
__global__ void k_edge_P(const int* __restrict__ ei,
                         const float* __restrict__ h,
                         const float* __restrict__ eall,
                         const float* __restrict__ W1, const float* __restrict__ b1,
                         const float* __restrict__ W2, const float* __restrict__ b2,
                         float* agg1, int nE) {
  __shared__ __align__(16) float lds[8][16 * LDS_COLS];
  int wid = threadIdx.x >> 5, lane = lane_id();
  int m = lane & 15, hi = lane >> 4;
  float* h48 = lds[wid];
  v16h B1[3], B2[2];
#pragma unroll
  for (int nb = 0; nb < 3; ++nb)
    B1[nb] = load_b_map(W1, 40, nb * 16, [](int k) -> int {
      if (k < 5) return k;
      if (k >= 8 && k < 13) return k - 3;
      if (k >= 16) return k - 6;   // rows 10..25
      return -1;
    });
  B2[0] = load_b_hidden(W2, 1, 0, 0);
  B2[1] = load_b_hidden(W2, 1, 32, 0);
  float bv1[3]; bias3(b1, bv1);
  float ob = bias1(b2, 1);
  int tiles = (nE + 15) >> 4;
  int gw = blockIdx.x * (blockDim.x >> 5) + wid;
  int nw = gridDim.x * (blockDim.x >> 5);
  for (int t = gw; t < tiles; t += nw) {
    int base = t << 4;
    int e0 = base + m; bool eok = e0 < nE; if (!eok) e0 = nE - 1;
    int s = ei[e0];
    int d = ei[(size_t)nE + e0];
    const float* hr = h + (size_t)(hi ? s : d) * 8;
    const float* er = eall + (size_t)e0 * 16 + hi * 8;
    float4 l0 = ld4(hr), l1 = ld4(hr + 4);
    float4 u0 = ld4(er), u1 = ld4(er + 4);
    layer_first(B1, bv1, h48, pack_a(l0, l1, u0, u1));
    v8f c = layer_last(B2, h48);
    stage_out(h48, c, ob, 1, 0);
    if (!hi && eok)
      unsafeAtomicAdd(&agg1[d], h48[m * LDS_COLS]);
  }
}

// P node model: padded [k0..7]=h, [k8]=agg1. 6->40->1 -> out.
__global__ void k_node_P(const float* __restrict__ h,
                         const float* __restrict__ agg1,
                         const float* __restrict__ W1, const float* __restrict__ b1,
                         const float* __restrict__ W2, const float* __restrict__ b2,
                         float* __restrict__ out, int nN) {
  __shared__ __align__(16) float lds[8][16 * LDS_COLS];
  int wid = threadIdx.x >> 5, lane = lane_id();
  int m = lane & 15, hi = lane >> 4;
  float* h48 = lds[wid];
  v16h B1[3], B2[2];
#pragma unroll
  for (int nb = 0; nb < 3; ++nb)
    B1[nb] = load_b_map(W1, 40, nb * 16, [](int k) -> int {
      if (k < 5) return k;
      if (k == 8) return 5;
      return -1;
    });
  B2[0] = load_b_hidden(W2, 1, 0, 0);
  B2[1] = load_b_hidden(W2, 1, 32, 0);
  float bv1[3]; bias3(b1, bv1);
  float ob = bias1(b2, 1);
  int tiles = (nN + 15) >> 4;
  int gw = blockIdx.x * (blockDim.x >> 5) + wid;
  int nw = gridDim.x * (blockDim.x >> 5);
  float4 z = {0.f, 0.f, 0.f, 0.f};
  for (int t = gw; t < tiles; t += nw) {
    int base = t << 4;
    int node = base + m; bool ok = node < nN; if (!ok) node = nN - 1;
    float av = agg1[node];                               // all lanes load (cheap)
    float4 h0 = ld4(h + (size_t)node * 8);
    float4 h1 = ld4(h + (size_t)node * 8 + 4);
    float4 l0 = hi ? (float4){av, 0.f, 0.f, 0.f} : h0;   // cndmask, no branch
    float4 l1 = hi ? z : h1;
    layer_first(B1, bv1, h48, pack_a(l0, l1, z, z));
    v8f c = layer_last(B2, h48);
    stage_out(h48, c, ob, 1, 0);
    if (!hi && ok) out[node] = h48[m * LDS_COLS];
  }
}

// 4-layer MLP head. mode 0: rows=edges, in[row*16+k], K=16 (W head).
//                   mode 1: rows=nodes, in[row*8+k],  K=5  (B / X heads).
__global__ void k_mlp4(const float* __restrict__ in, int mode,
                       const float* __restrict__ W1, const float* __restrict__ b1,
                       const float* __restrict__ W2, const float* __restrict__ b2,
                       const float* __restrict__ W3, const float* __restrict__ b3,
                       const float* __restrict__ W4, const float* __restrict__ b4,
                       int NOUT, int dosig, float* __restrict__ out, int nR) {
  __shared__ __align__(16) float lds[8][16 * LDS_COLS];
  int wid = threadIdx.x >> 5, lane = lane_id();
  int m = lane & 15, hi = lane >> 4;
  float sel = hi ? 0.0f : 1.0f;
  float* h48 = lds[wid];
  v16h B1[3], B2[6], B3[6], B4[2];
#pragma unroll
  for (int nb = 0; nb < 3; ++nb) {
    B1[nb] = load_b_map(W1, 40, nb * 16, [mode](int k) -> int {
      return (k < (mode == 0 ? 16 : 5)) ? k : -1;
    });
    B2[nb]     = load_b_hidden(W2, 40, 0,  nb * 16);
    B2[3 + nb] = load_b_hidden(W2, 40, 32, nb * 16);
    B3[nb]     = load_b_hidden(W3, 40, 0,  nb * 16);
    B3[3 + nb] = load_b_hidden(W3, 40, 32, nb * 16);
  }
  B4[0] = load_b_hidden(W4, NOUT, 0, 0);
  B4[1] = load_b_hidden(W4, NOUT, 32, 0);
  float bv1[3], bv2[3], bv3[3];
  bias3(b1, bv1); bias3(b2, bv2); bias3(b3, bv3);
  float ob = bias1(b4, NOUT);
  int tiles = (nR + 15) >> 4;
  int gw = blockIdx.x * (blockDim.x >> 5) + wid;
  int nw = gridDim.x * (blockDim.x >> 5);
  float4 z = {0.f, 0.f, 0.f, 0.f};
  for (int t = gw; t < tiles; t += nw) {
    int base = t << 4;
    int row = base + m; bool ok = row < nR; if (!ok) row = nR - 1;
    float4 l0, l1;
    if (mode == 0) {                                   // kernel-uniform branch
      const float* r = in + (size_t)row * 16 + hi * 8;
      l0 = ld4(r); l1 = ld4(r + 4);
    } else {
      const float* r = in + (size_t)row * 8;
      float4 t0 = ld4(r), t1 = ld4(r + 4);             // hi lanes: dup cacheline
      l0 = {t0.x * sel, t0.y * sel, t0.z * sel, t0.w * sel};
      l1 = {t1.x * sel, t1.y * sel, t1.z * sel, t1.w * sel};
    }
    layer_first(B1, bv1, h48, pack_a(l0, l1, z, z));
    layer_mid(B2, bv2, h48);
    layer_mid(B3, bv3, h48);
    v8f c = layer_last(B4, h48);
    stage_out(h48, c, ob, 2, 0);
    if (!hi && ok) {
      const float* orow = h48 + m * LDS_COLS;
      if (NOUT == 1) {
        float v = orow[0];
        if (dosig) v = sigmoidf(v);
        out[row] = v;
      } else {
        float2 v = {orow[0], orow[1]};
        if (dosig) { v.x = sigmoidf(v.x); v.y = sigmoidf(v.y); }
        *(float2*)&out[(size_t)row * 2] = v;
      }
    }
  }
}

// ---------------------------------------------------------------- host

extern "C" void kernel_launch(void* const* d_in, const int* in_sizes, int n_in,
                              void* d_out, int out_size, void* d_ws, size_t ws_size,
                              hipStream_t stream) {
  (void)out_size; (void)ws_size;
  const float* x  = (const float*)d_in[0];
  const int*   ei = (const int*)  d_in[1];
  const float* ea = (const float*)d_in[2];
  int N = in_sizes[0] / 14;
  int E = in_sizes[2] / 4;

  struct Lin { const float *w, *b; };
  Lin lin[48];
  int nl = 0;
  for (int i = 3; i + 1 < n_in && nl < 48; i += 2) {
    lin[nl].w = (const float*)d_in[i];
    lin[nl].b = (const float*)d_in[i + 1];
    ++nl;
  }
  // 0-1 ec_node_enc, 2-3 ec_edge_enc, 4-15 ec_layers (edgeL1,edgeL2,nodeL1,nodeL2 x3),
  // 16-17 hc_node_enc, 18-19 hc_edge_enc, 20-31 hc_layers, 32-35 W, 36-39 B,
  // 40-43 X, 44-45 P.edge, 46-47 P.node.

  float* ws   = (float*)d_ws;
  float* hA   = ws;
  float* hB   = hA + (size_t)N * 8;
  float* agg  = hB + (size_t)N * 8;
  float* eAll = agg + (size_t)N * 4;   // E*16, reused by EC then HC branch

  float* out_w     = (float*)d_out;          // [E]
  float* out_h     = out_w + (size_t)E;      // [N,2]
  float* out_beta  = out_h + (size_t)N * 2;  // [N]
  float* out_track = out_beta + (size_t)N;   // [N]

  dim3 blk(256);
  int gE = 2048, gN = 512;

  // ---- EC branch ----
  k_node_enc<<<gN, blk, 0, stream>>>(x, lin[0].w, lin[0].b, lin[1].w, lin[1].b, hA, N);
  k_edge_enc<<<gE, blk, 0, stream>>>(ea, (const float*)nullptr, 0,
                                     lin[2].w, lin[2].b, lin[3].w, lin[3].b, eAll, E);
  float* hc = hA; float* hn = hB;
  for (int l = 0; l < 3; ++l) {
    const Lin* L = &lin[4 + 4 * l];
    k_zero<<<(N * 4 + 255) / 256, blk, 0, stream>>>(agg, N * 4);
    k_edge_layer<<<gE, blk, 0, stream>>>(ei, hc, eAll, 4 * l, 4 * (l + 1),
                                         L[0].w, L[0].b, L[1].w, L[1].b, agg, E);
    k_node_layer<<<gN, blk, 0, stream>>>(hc, agg, L[2].w, L[2].b, L[3].w, L[3].b, hn, N);
    float* tmp = hc; hc = hn; hn = tmp;
  }
  k_mlp4<<<gE, blk, 0, stream>>>(eAll, 0,
                                 lin[32].w, lin[32].b, lin[33].w, lin[33].b,
                                 lin[34].w, lin[34].b, lin[35].w, lin[35].b,
                                 1, 1, out_w, E);

  // ---- HC branch ----
  k_node_enc<<<gN, blk, 0, stream>>>(x, lin[16].w, lin[16].b, lin[17].w, lin[17].b, hA, N);
  k_edge_enc<<<gE, blk, 0, stream>>>(ea, out_w, 1,
                                     lin[18].w, lin[18].b, lin[19].w, lin[19].b, eAll, E);
  hc = hA; hn = hB;
  for (int l = 0; l < 3; ++l) {
    const Lin* L = &lin[20 + 4 * l];
    k_zero<<<(N * 4 + 255) / 256, blk, 0, stream>>>(agg, N * 4);
    k_edge_layer<<<gE, blk, 0, stream>>>(ei, hc, eAll, 4 * l, 4 * (l + 1),
                                         L[0].w, L[0].b, L[1].w, L[1].b, agg, E);
    k_node_layer<<<gN, blk, 0, stream>>>(hc, agg, L[2].w, L[2].b, L[3].w, L[3].b, hn, N);
    float* tmp = hc; hc = hn; hn = tmp;
  }
  // hc = final h_hc
  k_mlp4<<<gN, blk, 0, stream>>>(hc, 1,
                                 lin[36].w, lin[36].b, lin[37].w, lin[37].b,
                                 lin[38].w, lin[38].b, lin[39].w, lin[39].b,
                                 1, 1, out_beta, N);   // beta = sigmoid(B(h_hc))
  k_mlp4<<<gN, blk, 0, stream>>>(hc, 1,
                                 lin[40].w, lin[40].b, lin[41].w, lin[41].b,
                                 lin[42].w, lin[42].b, lin[43].w, lin[43].b,
                                 2, 0, out_h, N);      // h = X(h_hc)
  k_zero<<<(N + 255) / 256, blk, 0, stream>>>(agg, N);
  k_edge_P<<<gE, blk, 0, stream>>>(ei, hc, eAll,
                                   lin[44].w, lin[44].b, lin[45].w, lin[45].b, agg, E);
  k_node_P<<<gN, blk, 0, stream>>>(hc, agg,
                                   lin[46].w, lin[46].b, lin[47].w, lin[47].b,
                                   out_track, N);
}
